// CaptionDecoderCell_20469814133167
// MI455X (gfx1250) — matrix operand order repaired and checked by reading, not compiled, not run on hardware
//
#include <hip/hip_runtime.h>

// Problem constants (fixed by the reference).
#define BB   256
#define EE   512
#define MM2  512
#define TT   20
#define VV   10000
#define K2E  1024   // 2*E
#define G4   2048   // 4*M2

typedef __attribute__((ext_vector_type(16))) __bf16 v16bf;
typedef __attribute__((ext_vector_type(8)))  __bf16 v8bf;
typedef __attribute__((ext_vector_type(8)))  float  v8f;

// ---------------- fp32 -> bf16 conversion (grid-stride) ----------------
__global__ void cvt_f32_bf16(const float* __restrict__ src, __bf16* __restrict__ dst, int n) {
    for (int i = blockIdx.x * blockDim.x + threadIdx.x; i < n; i += gridDim.x * blockDim.x)
        dst[i] = (__bf16)src[i];
}

// ---------------- build teacher-forced inputs X (T*B x 2E, bf16) -------
// row r = t*B + b : [ (t==0 ? o[b] : emb[tok[b, t-1]]) , o[b] ]
__global__ void build_X(const int* __restrict__ tok, const float* __restrict__ emb,
                        const float* __restrict__ o, __bf16* __restrict__ X) {
    const int r = blockIdx.x;
    const int ts = r / BB;
    const int b  = r % BB;
    const float* first = (ts == 0) ? (o + (size_t)b * EE)
                                   : (emb + (size_t)tok[b * TT + (ts - 1)] * EE);
    const float* second = o + (size_t)b * EE;
    __bf16* row = X + (size_t)r * K2E;
    #pragma unroll
    for (int k = 0; k < 4; ++k) {
        int e = threadIdx.x + k * 256;
        row[e] = (e < EE) ? (__bf16)first[e] : (__bf16)second[e - EE];
    }
}

// ---------------- bf16 WMMA GEMM: out(MxN) = A(MxK) * W(NxK)^T + bias --
// MODE 0: out[m*ldc + n]  (+= addsrc[m*ldc+n] when HAS_ADD)
// MODE 1: logits layout  out[b*(V*T) + n*T + t], with m = t*B + b
template <int MODE, bool HAS_ADD>
__global__ __launch_bounds__(256)
void gemm_bf16(const __bf16* __restrict__ A, int lda,
               const __bf16* __restrict__ W, int ldw,
               const float* __restrict__ bias,
               const float* __restrict__ addsrc,
               float* __restrict__ out, int ldc,
               int N, int K) {
    const int tid  = threadIdx.x;
    const int wv   = tid >> 5;
    const int lane = tid & 31;
    const int ln   = lane & 15;
    const int kh   = lane >> 4;           // 0/1 : K-half of the wave
    const int wm   = wv >> 2;             // 0..1
    const int wn   = wv & 3;              // 0..3
    const int m0   = blockIdx.x * 64 + wm * 32;
    const int n0   = blockIdx.y * 256 + wn * 64;

    v8f acc[2][4];
    #pragma unroll
    for (int i = 0; i < 2; ++i)
        #pragma unroll
        for (int j = 0; j < 4; ++j)
            #pragma unroll
            for (int r = 0; r < 8; ++r) acc[i][j][r] = 0.0f;

    for (int kb = 0; kb < K; kb += 32) {
        v16bf a[2], b[4];
        #pragma unroll
        for (int i = 0; i < 2; ++i) {
            // A 16x32 bf16 layout: lane<16 -> K {0..7,16..23}; lane>=16 -> +8
            const __bf16* ap = A + (size_t)(m0 + i * 16 + ln) * lda + kb + kh * 8;
            ((v8bf*)&a[i])[0] = *(const v8bf*)ap;
            ((v8bf*)&a[i])[1] = *(const v8bf*)(ap + 16);
        }
        #pragma unroll
        for (int j = 0; j < 4; ++j) {
            // B 32x16 bf16 layout: lane gives column n, K {0..7,16..23} (+8 for lanes>=16)
            int n = n0 + j * 16 + ln;
            if (n > N - 1) n = N - 1;          // clamp (EXEC must stay all-ones)
            const __bf16* bp = W + (size_t)n * ldw + kb + kh * 8;
            ((v8bf*)&b[j])[0] = *(const v8bf*)bp;
            ((v8bf*)&b[j])[1] = *(const v8bf*)(bp + 16);
        }
        #pragma unroll
        for (int i = 0; i < 2; ++i)
            #pragma unroll
            for (int j = 0; j < 4; ++j)
                acc[i][j] = __builtin_amdgcn_wmma_f32_16x16x32_bf16(
                    false, a[i], false, b[j], (short)0, acc[i][j], false, false);
    }

    // Epilogue. C/D layout: lane -> n = ln ; VGPR r -> m = kh*8 + r
    #pragma unroll
    for (int i = 0; i < 2; ++i) {
        #pragma unroll
        for (int j = 0; j < 4; ++j) {
            const int n = n0 + j * 16 + ln;
            if (n >= N) continue;
            const float bs = bias[n];
            #pragma unroll
            for (int r = 0; r < 8; ++r) {
                const int m = m0 + i * 16 + kh * 8 + r;
                float v = acc[i][j][r] + bs;
                if (HAS_ADD) v += addsrc[(size_t)m * ldc + n];
                if (MODE == 0) {
                    out[(size_t)m * ldc + n] = v;
                } else {
                    const int tt = m >> 8;     // m / 256  (B == 256)
                    const int bb = m & 255;
                    out[(size_t)bb * (VV * TT) + (size_t)n * TT + tt] = v;
                }
            }
        }
    }
}

// ---------------- pointwise LSTM cell update ----------------------------
__global__ void lstm_pointwise(const float* __restrict__ gates,  // (B, 4*M2)
                               float* __restrict__ cbuf,         // (B, M2)
                               float* __restrict__ hf32,         // (B, M2)
                               __bf16* __restrict__ hbf) {       // (B, M2)
    const int idx = blockIdx.x * blockDim.x + threadIdx.x;       // b*M2 + n
    const int b = idx / MM2;
    const int n = idx % MM2;
    const float* g = gates + (size_t)b * G4;
    const float gi = g[n];
    const float gf = g[MM2 + n];
    const float gg = g[2 * MM2 + n];
    const float go = g[3 * MM2 + n];
    const float i = 1.0f / (1.0f + __expf(-gi));
    const float f = 1.0f / (1.0f + __expf(-gf));
    const float og = 1.0f / (1.0f + __expf(-go));
    const float gt = tanhf(gg);
    const float c = f * cbuf[idx] + i * gt;
    const float h = og * tanhf(c);
    cbuf[idx] = c;
    hf32[idx] = h;
    hbf[idx]  = (__bf16)h;
}

// ---------------- gather hidden state at length-1 -----------------------
__global__ void gather_hid(const int* __restrict__ length,
                           const float* __restrict__ Hf32,   // (T, B, M2)
                           float* __restrict__ out) {        // (B, M2)
    const int idx = blockIdx.x * blockDim.x + threadIdx.x;   // b*M2 + n
    const int b = idx / MM2;
    int s = length[b] - 1;
    if (s < 0) s = 0;
    out[idx] = Hf32[(size_t)s * BB * MM2 + idx];
}

extern "C" void kernel_launch(void* const* d_in, const int* in_sizes, int n_in,
                              void* d_out, int out_size, void* d_ws, size_t ws_size,
                              hipStream_t stream) {
    (void)in_sizes; (void)n_in; (void)out_size; (void)ws_size;
    const float* o      = (const float*)d_in[0];
    const int*   tok    = (const int*)  d_in[1];
    const int*   length = (const int*)  d_in[2];
    const float* Wh     = (const float*)d_in[3];
    const float* bh     = (const float*)d_in[4];
    const float* Wc     = (const float*)d_in[5];
    const float* bc     = (const float*)d_in[6];
    const float* emb    = (const float*)d_in[7];
    const float* W_ih   = (const float*)d_in[8];
    const float* b_ih   = (const float*)d_in[9];
    const float* W_hh   = (const float*)d_in[10];
    const float* b_hh   = (const float*)d_in[11];
    const float* Wl     = (const float*)d_in[12];
    const float* bl     = (const float*)d_in[13];
    float* out = (float*)d_out;

    // Workspace carve-up (256B aligned chunks).
    char* ws = (char*)d_ws;
    size_t off = 0;
    auto carve = [&](size_t bytes) { char* p = ws + off; off += (bytes + 255) & ~(size_t)255; return p; };
    __bf16* o_bf   = (__bf16*)carve((size_t)BB * EE * 2);
    __bf16* Wh_bf  = (__bf16*)carve((size_t)MM2 * EE * 2);
    __bf16* Wc_bf  = (__bf16*)carve((size_t)MM2 * EE * 2);
    __bf16* Wih_bf = (__bf16*)carve((size_t)G4 * K2E * 2);
    __bf16* Whh_bf = (__bf16*)carve((size_t)G4 * MM2 * 2);
    __bf16* Wl_bf  = (__bf16*)carve((size_t)VV * MM2 * 2);
    __bf16* X_bf   = (__bf16*)carve((size_t)TT * BB * K2E * 2);
    float*  gates  = (float*) carve((size_t)TT * BB * G4 * 4);
    float*  h0f    = (float*) carve((size_t)BB * MM2 * 4);
    float*  cbuf   = (float*) carve((size_t)BB * MM2 * 4);
    __bf16* h0bf   = (__bf16*)carve((size_t)BB * MM2 * 2);
    float*  Hf32   = (float*) carve((size_t)TT * BB * MM2 * 4);
    __bf16* Hbf    = (__bf16*)carve((size_t)TT * BB * MM2 * 2);

    // 1) weight / input conversions to bf16
    cvt_f32_bf16<<<512, 256, 0, stream>>>(o,    o_bf,   BB * EE);
    cvt_f32_bf16<<<512, 256, 0, stream>>>(Wh,   Wh_bf,  MM2 * EE);
    cvt_f32_bf16<<<512, 256, 0, stream>>>(Wc,   Wc_bf,  MM2 * EE);
    cvt_f32_bf16<<<1024, 256, 0, stream>>>(W_ih, Wih_bf, G4 * K2E);
    cvt_f32_bf16<<<1024, 256, 0, stream>>>(W_hh, Whh_bf, G4 * MM2);
    cvt_f32_bf16<<<2048, 256, 0, stream>>>(Wl,   Wl_bf,  VV * MM2);

    // 2) build teacher-forced input rows
    build_X<<<TT * BB, 256, 0, stream>>>(tok, emb, o, X_bf);

    // 3) all input-gate preactivations in one GEMM: gates = X @ W_ih^T + b_ih
    gemm_bf16<0, false><<<dim3(TT * BB / 64, G4 / 256), 256, 0, stream>>>(
        X_bf, K2E, Wih_bf, K2E, b_ih, nullptr, gates, G4, G4, K2E);

    // 4) h0 / c0 projections
    gemm_bf16<0, false><<<dim3(BB / 64, MM2 / 256), 256, 0, stream>>>(
        o_bf, EE, Wh_bf, EE, bh, nullptr, h0f, MM2, MM2, EE);
    gemm_bf16<0, false><<<dim3(BB / 64, MM2 / 256), 256, 0, stream>>>(
        o_bf, EE, Wc_bf, EE, bc, nullptr, cbuf, MM2, MM2, EE);
    cvt_f32_bf16<<<256, 256, 0, stream>>>(h0f, h0bf, BB * MM2);

    // 5) sequential recurrence: gates[t] += h_{t-1} @ W_hh^T + b_hh ; pointwise
    for (int t = 0; t < TT; ++t) {
        const __bf16* hprev = (t == 0) ? h0bf : (Hbf + (size_t)(t - 1) * BB * MM2);
        float* gslice = gates + (size_t)t * BB * G4;
        gemm_bf16<0, true><<<dim3(BB / 64, G4 / 256), 256, 0, stream>>>(
            hprev, MM2, Whh_bf, MM2, b_hh, gslice, gslice, G4, G4, MM2);
        lstm_pointwise<<<BB * MM2 / 256, 256, 0, stream>>>(
            gslice, cbuf, Hf32 + (size_t)t * BB * MM2, Hbf + (size_t)t * BB * MM2);
    }

    // 6) batched logits GEMM with (B, V, T) transposed store
    gemm_bf16<1, false><<<dim3(TT * BB / 64, (VV + 255) / 256), 256, 0, stream>>>(
        Hbf, MM2, Wl_bf, MM2, bl, nullptr, out, 0, VV, MM2);

    // 7) gather hid at clip(length-1, 0)
    gather_hid<<<BB * MM2 / 256, 256, 0, stream>>>(
        length, Hf32, out + (size_t)BB * VV * TT);
}
